// MiniTransformer_3762391351596
// MI455X (gfx1250) — compile-verified, matrix operand order checked by reading, not run on hardware
//
#include <hip/hip_runtime.h>
#include <hip/hip_bf16.h>
#include <cstdint>
#include <cstddef>

typedef unsigned short u16;
typedef __attribute__((ext_vector_type(16))) __bf16 v16bf;
typedef __attribute__((ext_vector_type(8)))  float  v8f;

union FragAB { v16bf v; uint4 q[2]; };

__device__ __forceinline__ u16 f32_to_bf16(float f) {
  union { float f; uint32_t u; } cv; cv.f = f;
  uint32_t u = cv.u;
  u += 0x7FFFu + ((u >> 16) & 1u);   // round-to-nearest-even
  return (u16)(u >> 16);
}

__device__ __forceinline__ void store_val(float* p, float v) { *p = v; }
__device__ __forceinline__ void store_val(u16*   p, float v) { *p = f32_to_bf16(v); }

__device__ __forceinline__ float wave_sum(float v) {
  #pragma unroll
  for (int o = 16; o > 0; o >>= 1) v += __shfl_xor(v, o, 32);
  return v;
}
__device__ __forceinline__ float wave_max(float v) {
  #pragma unroll
  for (int o = 16; o > 0; o >>= 1) v = fmaxf(v, __shfl_xor(v, o, 32));
  return v;
}

// ---------------------------------------------------------------------------
// Tensor Data Mover staging (gfx1250): one wave issues a 2D tile DMA
// global->LDS; TDM applies LDS padding so rows land at an 80B stride.
// This toolchain's builtin takes 6 args (g0 x4, g1 x8, g2 x4, g3 x4, x8, cpol).
// ---------------------------------------------------------------------------
#if defined(__AMDGCN__) && __has_builtin(__builtin_amdgcn_tensor_load_to_lds)
#define USE_TDM 1
#else
#define USE_TDM 0
#endif

#define BM 128
#define BN 128
#define BKC 32
#define LSTR 40   // padded LDS row stride in u16 (80B, 16B aligned)

#if USE_TDM
typedef unsigned int v4u __attribute__((ext_vector_type(4)));
typedef int v8i __attribute__((ext_vector_type(8)));
typedef int v4i __attribute__((ext_vector_type(4)));

// Load a 128-row x 32-col bf16 tile whose top-left element is at gptr; the
// source matrix has leading dimension `ld` (elements) and `rows` total rows.
__device__ __forceinline__ void tdm_tile_load(const u16* gptr, unsigned lds_off,
                                              int rows, int ld) {
  unsigned long long ga = (unsigned long long)(uintptr_t)gptr;
  v4u g0;
  g0[0] = 1u;                                   // count=1, user mode
  g0[1] = lds_off;                              // lds_addr (bytes)
  g0[2] = (unsigned)ga;                         // global_addr[31:0]
  g0[3] = (unsigned)((ga >> 32) & 0x01FFFFFFu)  // global_addr[56:32]
        | (2u << 30);                           // type = 2 ("image")
  v8i g1;
  g1[0] = (1 << 16)   // data_size = 2 bytes
        | (1 << 20)   // pad_enable
        | (3 << 22)   // pad_interval code 3 = 16 DWORDs (= one 64B tile row)
        | (3 << 25);  // pad_amount  code 3 = 4 DWORDs (16B) -> 80B row stride
  g1[1] = (int)((unsigned)ld << 16);                                   // tensor_dim0[15:0]
  g1[2] = (int)((((unsigned)ld >> 16) & 0xFFFFu) |
                (((unsigned)rows & 0xFFFFu) << 16));                   // dim0 hi | dim1 lo
  g1[3] = (int)((((unsigned)rows >> 16) & 0xFFFFu) | (32u << 16));     // dim1 hi | tile_dim0=32
  g1[4] = 128;                                                         // tile_dim1=128
  g1[5] = ld;                                                          // dim0_stride[31:0]
  g1[6] = 0;
  g1[7] = 0;
  v4i zero4 = (v4i)0;
  v8i zero8 = (v8i)0;
  __builtin_amdgcn_tensor_load_to_lds(g0, g1, zero4, zero4, zero8, 0);
}
#endif

// ---------------------------------------------------------------------------
// fp32 -> bf16 conversion (vectorized by 4)
// ---------------------------------------------------------------------------
__global__ void __launch_bounds__(256)
cvt_f32_to_bf16(const float* __restrict__ in, u16* __restrict__ out, long long n) {
  long long i = ((long long)blockIdx.x * blockDim.x + threadIdx.x) * 4;
  if (i + 3 < n) {
    float4 f = *(const float4*)(in + i);
    uint2 pack;
    u16* pp = (u16*)&pack;
    pp[0] = f32_to_bf16(f.x); pp[1] = f32_to_bf16(f.y);
    pp[2] = f32_to_bf16(f.z); pp[3] = f32_to_bf16(f.w);
    *(uint2*)(out + i) = pack;
  }
}

// ---------------------------------------------------------------------------
// bf16 WMMA GEMM: C[M,N] = A[M,K] * Bsrc^T where Bsrc is [N,K] row-major
// (torch Linear weights; k for q@k^T; pre-transposed v for attn@v).
//   EPI: 0 = none, 1 = scale, 2 = +bias, 3 = +bias then ReLU
//   CT : store C transposed in [M/2048][N][2048] bf16 layout (packed b128)
// Block tile 128x128, K-chunk 32, 256 threads = 8 waves, wave tile 64x32.
// Double-buffered LDS fed by the Tensor Data Mover, one barrier per K-step.
// ---------------------------------------------------------------------------
template<int EPI, bool CT, typename OutT>
__global__ void __launch_bounds__(256)
gemm_bf16_wmma(const u16* __restrict__ A, const u16* __restrict__ Bsrc,
               OutT* __restrict__ C, const float* __restrict__ bias,
               int M, int N, int K, int lda, int ldb, int ldc,
               long long sA, long long sB, long long sC, float scale)
{
  __shared__ u16 As[2][BM * LSTR];
  __shared__ u16 Bs[2][BN * LSTR];

  const int z = blockIdx.z;
  A    += (long long)z * sA;
  Bsrc += (long long)z * sB;
  C    += (long long)z * sC;

  const int n0 = blockIdx.x * BN;
  const int m0 = blockIdx.y * BM;

  const int t    = threadIdx.x;
  const int lane = t & 31;
  const int wave = t >> 5;            // 0..7
  const int wm   = (wave >> 2) * 64;  // 0 or 64
  const int wn   = (wave & 3) * 32;   // 0,32,64,96

  const int l15 = lane & 15;
  const int hi  = lane >> 4;

  v8f acc[4][2];
  #pragma unroll
  for (int i = 0; i < 4; ++i)
    #pragma unroll
    for (int j = 0; j < 2; ++j)
      #pragma unroll
      for (int e = 0; e < 8; ++e) acc[i][j][e] = 0.0f;

  const u16* Atile = A    + (long long)m0 * lda;
  const u16* Btile = Bsrc + (long long)n0 * ldb;
  const int nK = K / BKC;

#if USE_TDM
  if (wave == 0) {
    tdm_tile_load(Atile, (unsigned)(uintptr_t)(void*)&As[0][0], M, lda);
    tdm_tile_load(Btile, (unsigned)(uintptr_t)(void*)&Bs[0][0], N, ldb);
  }
#endif

  for (int it = 0; it < nK; ++it) {
    const int buf = it & 1;
#if USE_TDM
    if (wave == 0) __builtin_amdgcn_s_wait_tensorcnt((short)0);
    __syncthreads();                       // publish TDM-filled buf to all waves
    if (wave == 0 && it + 1 < nK) {        // prefetch next K tile via TDM
      tdm_tile_load(Atile + (it + 1) * BKC,
                    (unsigned)(uintptr_t)(void*)&As[buf ^ 1][0], M, lda);
      tdm_tile_load(Btile + (it + 1) * BKC,
                    (unsigned)(uintptr_t)(void*)&Bs[buf ^ 1][0], N, ldb);
    }
#else
    // Fallback: cooperative vector copy (both toolchain variants compile this)
    #pragma unroll
    for (int c = t; c < (BM * BKC / 8); c += 256) {
      int r = c >> 2, col = (c & 3) * 8;   // 4 x 16B chunks per 32-elem row
      *(uint4*)&As[buf][r * LSTR + col] =
          *(const uint4*)(Atile + (long long)r * lda + it * BKC + col);
    }
    #pragma unroll
    for (int c = t; c < (BN * BKC / 8); c += 256) {
      int r = c >> 2, col = (c & 3) * 8;
      *(uint4*)&Bs[buf][r * LSTR + col] =
          *(const uint4*)(Btile + (long long)r * ldb + it * BKC + col);
    }
    __syncthreads();
#endif

    // ---- compute one K=32 WMMA step ----
    const int kb = hi * 8;   // lanes 0-15: K {0..7,16..23}; lanes 16-31: {8..15,24..31}
    FragAB a[4], b[2];
    #pragma unroll
    for (int i = 0; i < 4; ++i) {
      const u16* p = &As[buf][(wm + i * 16 + l15) * LSTR + kb];
      a[i].q[0] = *(const uint4*)p;
      a[i].q[1] = *(const uint4*)(p + 16);
    }
    #pragma unroll
    for (int j = 0; j < 2; ++j) {
      const u16* p = &Bs[buf][(wn + j * 16 + l15) * LSTR + kb];
      b[j].q[0] = *(const uint4*)p;
      b[j].q[1] = *(const uint4*)(p + 16);
    }
    #pragma unroll
    for (int i = 0; i < 4; ++i)
      #pragma unroll
      for (int j = 0; j < 2; ++j)
        acc[i][j] = __builtin_amdgcn_wmma_f32_16x16x32_bf16(
            false, a[i].v, false, b[j].v, (short)0, acc[i][j], false, false);

#if !USE_TDM
    __syncthreads();
#endif
  }

  // ---- epilogue: C/D layout — lane l15 = N column, vgpr e -> M = hi*8 + e ----
  #pragma unroll
  for (int j = 0; j < 2; ++j) {
    const int gn = n0 + wn + j * 16 + l15;
    float bv = 0.0f;
    if (EPI >= 2) bv = bias[gn];
    #pragma unroll
    for (int i = 0; i < 4; ++i) {
      const int gmb = m0 + wm + i * 16 + hi * 8;
      if (CT) {
        // transposed bf16 store: [gmb>>11][gn][gmb&2047], 8 contiguous elems
        uint4 pack;
        u16* pp = (u16*)&pack;
        #pragma unroll
        for (int e = 0; e < 8; ++e) pp[e] = f32_to_bf16(acc[i][j][e]);
        long long off = ((long long)(gmb >> 11) * N + gn) * 2048 + (gmb & 2047);
        *(uint4*)((u16*)C + off) = pack;
      } else {
        #pragma unroll
        for (int e = 0; e < 8; ++e) {
          float v = acc[i][j][e];
          if (EPI == 1) v *= scale;
          if (EPI >= 2) v += bv;
          if (EPI == 3) v = fmaxf(v, 0.0f);
          store_val(C + (long long)(gmb + e) * ldc + gn, v);
        }
      }
    }
  }
}

// ---------------------------------------------------------------------------
// Row softmax: one block per 2048-wide row; fp32 in, bf16 probs out
// ---------------------------------------------------------------------------
__global__ void __launch_bounds__(256)
softmax_rows(const float* __restrict__ scores, u16* __restrict__ probs) {
  const long long row = blockIdx.x;
  const float* r = scores + row * 2048;
  u16* o = probs + row * 2048;
  const int t = threadIdx.x, lane = t & 31, wv = t >> 5;
  __shared__ float red[8];

  float4 f0 = *(const float4*)(r + t * 8);
  float4 f1 = *(const float4*)(r + t * 8 + 4);
  float v[8] = {f0.x, f0.y, f0.z, f0.w, f1.x, f1.y, f1.z, f1.w};

  float m = v[0];
  #pragma unroll
  for (int i = 1; i < 8; ++i) m = fmaxf(m, v[i]);
  m = wave_max(m);
  if (lane == 0) red[wv] = m;
  __syncthreads();
  if (t == 0) { float x = red[0]; for (int i = 1; i < 8; ++i) x = fmaxf(x, red[i]); red[0] = x; }
  __syncthreads();
  const float rowmax = red[0];
  __syncthreads();

  float s = 0.0f;
  #pragma unroll
  for (int i = 0; i < 8; ++i) { v[i] = __expf(v[i] - rowmax); s += v[i]; }
  s = wave_sum(s);
  if (lane == 0) red[wv] = s;
  __syncthreads();
  if (t == 0) { float x = 0.0f; for (int i = 0; i < 8; ++i) x += red[i]; red[0] = x; }
  __syncthreads();
  const float inv = 1.0f / red[0];

  uint4 pack;
  u16* pp = (u16*)&pack;
  #pragma unroll
  for (int i = 0; i < 8; ++i) pp[i] = f32_to_bf16(v[i] * inv);
  *(uint4*)(o + t * 8) = pack;
}

// ---------------------------------------------------------------------------
// out = LayerNorm(a + b) * gamma + beta ; optional bf16 copy of the output
// one block per row of D=1024, 256 threads x 4 elements
// ---------------------------------------------------------------------------
__global__ void __launch_bounds__(256)
add_layernorm(const float* __restrict__ a, const float* __restrict__ b,
              const float* __restrict__ gamma, const float* __restrict__ beta,
              float* __restrict__ outf, u16* __restrict__ outb, float eps) {
  const long long row = blockIdx.x;
  const int t = threadIdx.x, lane = t & 31, wv = t >> 5;
  const float* pa = a + row * 1024;
  const float* pb = b + row * 1024;
  __shared__ float red[8];

  float4 fa = *(const float4*)(pa + t * 4);
  float4 fb = *(const float4*)(pb + t * 4);
  float v[4] = {fa.x + fb.x, fa.y + fb.y, fa.z + fb.z, fa.w + fb.w};

  float s = v[0] + v[1] + v[2] + v[3];
  s = wave_sum(s);
  if (lane == 0) red[wv] = s;
  __syncthreads();
  if (t == 0) { float x = 0.0f; for (int i = 0; i < 8; ++i) x += red[i]; red[0] = x; }
  __syncthreads();
  const float mu = red[0] * (1.0f / 1024.0f);
  __syncthreads();

  float ss = 0.0f;
  #pragma unroll
  for (int i = 0; i < 4; ++i) { float d = v[i] - mu; ss += d * d; }
  ss = wave_sum(ss);
  if (lane == 0) red[wv] = ss;
  __syncthreads();
  if (t == 0) { float x = 0.0f; for (int i = 0; i < 8; ++i) x += red[i]; red[0] = x; }
  __syncthreads();
  const float rs = rsqrtf(red[0] * (1.0f / 1024.0f) + eps);

  float4 g = *(const float4*)(gamma + t * 4);
  float4 be = *(const float4*)(beta + t * 4);
  float4 of;
  of.x = (v[0] - mu) * rs * g.x + be.x;
  of.y = (v[1] - mu) * rs * g.y + be.y;
  of.z = (v[2] - mu) * rs * g.z + be.z;
  of.w = (v[3] - mu) * rs * g.w + be.w;
  *(float4*)(outf + row * 1024 + t * 4) = of;

  if (outb) {
    uint2 pack;
    u16* pp = (u16*)&pack;
    pp[0] = f32_to_bf16(of.x); pp[1] = f32_to_bf16(of.y);
    pp[2] = f32_to_bf16(of.z); pp[3] = f32_to_bf16(of.w);
    *(uint2*)(outb + row * 1024 + t * 4) = pack;
  }
}

// ---------------------------------------------------------------------------
// Host launcher
// ---------------------------------------------------------------------------
extern "C" void kernel_launch(void* const* d_in, const int* in_sizes, int n_in,
                              void* d_out, int out_size, void* d_ws, size_t ws_size,
                              hipStream_t stream) {
  (void)in_sizes; (void)n_in; (void)out_size; (void)ws_size;

  const long long Bb = 8, S = 2048, D = 1024, H = 2048;
  const long long nTok = Bb * S;              // 16384

  const float* x   = (const float*)d_in[0];
  const float* Wq  = (const float*)d_in[1];
  const float* Wk  = (const float*)d_in[2];
  const float* Wv  = (const float*)d_in[3];
  const float* W1  = (const float*)d_in[4];
  const float* b1  = (const float*)d_in[5];
  const float* W2  = (const float*)d_in[6];
  const float* b2  = (const float*)d_in[7];
  const float* g1  = (const float*)d_in[8];
  const float* be1 = (const float*)d_in[9];
  const float* g2  = (const float*)d_in[10];
  const float* be2 = (const float*)d_in[11];
  float* out = (float*)d_out;

  // workspace layout (aliased across phases)
  char* p = (char*)d_ws;
  u16* xb  = (u16*)p; p += nTok * D * 2;
  u16* qb  = (u16*)p; p += nTok * D * 2;   // later reused as bf16 h
  u16* kb  = (u16*)p; p += nTok * D * 2;
  u16* vTb = (u16*)p; p += nTok * D * 2;   // v stored transposed: [b][D][S]
  u16* Wqb = (u16*)p; p += D * D * 2;
  u16* Wkb = (u16*)p; p += D * D * 2;
  u16* Wvb = (u16*)p; p += D * D * 2;
  u16* W1b = (u16*)p; p += H * D * 2;
  u16* W2b = (u16*)p; p += D * H * 2;
  float* scores = (float*)p; p += Bb * S * S * 4;  // reused: attn_out f32, then ff2 f32
  u16*   probs  = (u16*)p;   p += Bb * S * S * 2;  // reused: ff1 bf16
  float* h      = (float*)p; p += nTok * D * 4;

  float* attn_out = scores;
  u16*   hb       = qb;
  u16*   ff1      = probs;
  float* ff2      = scores;

  const dim3 blk(256);

  // ---- fp32 -> bf16 conversions ----
  auto cvt = [&](const float* src, u16* dst, long long n) {
    long long nt = n / 4;
    cvt_f32_to_bf16<<<dim3((unsigned)((nt + 255) / 256)), blk, 0, stream>>>(src, dst, n);
  };
  cvt(x,  xb,  nTok * D);
  cvt(Wq, Wqb, D * D);
  cvt(Wk, Wkb, D * D);
  cvt(Wv, Wvb, D * D);
  cvt(W1, W1b, H * D);
  cvt(W2, W2b, D * H);

  // ---- q/k projections [16384,1024]; v projection stored transposed ----
  {
    dim3 g(8, 128, 1);
    gemm_bf16_wmma<0, false, u16><<<g, blk, 0, stream>>>(
        xb, Wqb, qb, nullptr, (int)nTok, (int)D, (int)D, (int)D, (int)D, (int)D,
        0, 0, 0, 1.0f);
    gemm_bf16_wmma<0, false, u16><<<g, blk, 0, stream>>>(
        xb, Wkb, kb, nullptr, (int)nTok, (int)D, (int)D, (int)D, (int)D, (int)D,
        0, 0, 0, 1.0f);
    gemm_bf16_wmma<0, true, u16><<<g, blk, 0, stream>>>(
        xb, Wvb, vTb, nullptr, (int)nTok, (int)D, (int)D, (int)D, (int)D, (int)D,
        0, 0, 0, 1.0f);
  }

  // ---- scores[b] = q[b] @ k[b]^T / sqrt(D) : per-batch [2048,2048] ----
  {
    dim3 g(16, 16, 8);
    gemm_bf16_wmma<1, false, float><<<g, blk, 0, stream>>>(
        qb, kb, scores, nullptr, (int)S, (int)S, (int)D, (int)D, (int)D, (int)S,
        S * D, S * D, S * S, 0.03125f /* 1/sqrt(1024) */);
  }

  // ---- softmax over last dim, bf16 probs ----
  softmax_rows<<<dim3((unsigned)(Bb * S)), blk, 0, stream>>>(scores, probs);

  // ---- attn_out[b] = probs[b] @ vT[b]^T : [2048,1024] (vT is [D][S]) ----
  {
    dim3 g(8, 16, 8);
    gemm_bf16_wmma<0, false, float><<<g, blk, 0, stream>>>(
        probs, vTb, attn_out, nullptr, (int)S, (int)D, (int)S, (int)S, (int)S, (int)D,
        S * S, D * S, S * D, 1.0f);
  }

  // ---- h = LN(x + attn_out); also bf16 copy for FFN ----
  add_layernorm<<<dim3((unsigned)nTok), blk, 0, stream>>>(
      x, attn_out, g1, be1, h, hb, 1e-5f);

  // ---- ff1 = relu(h @ W1^T + b1) : [16384,2048] bf16 ----
  {
    dim3 g(16, 128, 1);
    gemm_bf16_wmma<3, false, u16><<<g, blk, 0, stream>>>(
        hb, W1b, ff1, b1, (int)nTok, (int)H, (int)D, (int)D, (int)D, (int)H,
        0, 0, 0, 1.0f);
  }

  // ---- ff2 = ff1 @ W2^T + b2 : [16384,1024] f32 ----
  {
    dim3 g(8, 128, 1);
    gemm_bf16_wmma<2, false, float><<<g, blk, 0, stream>>>(
        ff1, W2b, ff2, b2, (int)nTok, (int)D, (int)H, (int)H, (int)H, (int)D,
        0, 0, 0, 1.0f);
  }

  // ---- out = LN(h + ff2) ----
  add_layernorm<<<dim3((unsigned)nTok), blk, 0, stream>>>(
      h, ff2, g2, be2, out, nullptr, 1e-5f);
}
